// Attention_26809185862242
// MI455X (gfx1250) — compile-verified, matrix operand order checked by reading, not run on hardware
//
#include <hip/hip_runtime.h>
#include <hip/hip_bf16.h>

typedef __bf16 bf16;
typedef __attribute__((ext_vector_type(8)))  __bf16 bf16x8;
typedef __attribute__((ext_vector_type(16))) __bf16 v16bf;
typedef __attribute__((ext_vector_type(8)))  float  v8f;
typedef __attribute__((ext_vector_type(4)))  float  f32x4;

#define MODEL_DIMS 768
#define CHANNELS   2048
#define N_HEADS    16
#define HEAD_DIM   128
#define SEQ        4096
#define BATCH      4
#define SCALE      0.08838834764831845f   // 128^-0.5

union FragU { v16bf v; bf16x8 h[2]; };

// 16x32 bf16 fragment (A layout, or B supplied as NxK) from a row-major tile
// with row stride ldk elements.  Lanes 0-15 hold K {0..7,16..23}, lanes 16-31
// hold K {8..15,24..31}; lane row = lane & 15.  Two b128 loads per lane.
__device__ __forceinline__ v16bf load_frag(const bf16* base, int ldk) {
  const int lane = threadIdx.x & 31;
  const int m  = lane & 15;
  const int hf = lane >> 4;
  FragU f;
  f.h[0] = *(const bf16x8*)(base + (size_t)m * ldk + hf * 8);
  f.h[1] = *(const bf16x8*)(base + (size_t)m * ldk + 16 + hf * 8);
  return f.v;
}

__device__ __forceinline__ v8f wmma_bf16(v16bf a, v16bf b, v8f c) {
  return __builtin_amdgcn_wmma_f32_16x16x32_bf16(
      /*neg_a=*/false, a, /*neg_b=*/false, b,
      /*c_mod=*/(short)0, c, /*reuse_a=*/false, /*reuse_b=*/false);
}

// Async global->LDS copy (CDNA5 Tensor/Async path, tracked by ASYNCcnt).
// The LDS operand is the low 32 bits of the flat pointer (aperture|offset).
__device__ __forceinline__ void async_load_b128(void* lds_dst,
                                                const void* gsrc) {
  asm volatile("global_load_async_to_lds_b128 %0, %1, off"
               :
               : "v"((unsigned)(unsigned long long)lds_dst), "v"(gsrc)
               : "memory");
}
__device__ __forceinline__ void wait_async0() {
  asm volatile("s_wait_asynccnt 0" ::: "memory");
}

// ---------------------------------------------------------------------------
// Kernel 1: fused QKV projection.  X[16384x768] @ W[768x2048] + b, fp32 in,
// bf16 out, scattered into per-head layout [B][H][SEQ][HEAD_DIM].
// Block: 256 threads (8 waves), tile 128(M) x 128(N); wave = 32x64 sub-tile.
// A/B staging needs fp32->bf16 conversion, so it stays on the VALU path.
// ---------------------------------------------------------------------------
__global__ __launch_bounds__(256) void qkv_proj_kernel(
    const float* __restrict__ q, const float* __restrict__ k,
    const float* __restrict__ v,
    const float* __restrict__ Wq, const float* __restrict__ bq,
    const float* __restrict__ Wk, const float* __restrict__ bk,
    const float* __restrict__ Wv, const float* __restrict__ bv,
    bf16* __restrict__ Qh, bf16* __restrict__ Kh, bf16* __restrict__ Vh) {
  const float *X, *W, *bias;
  bf16* outp;
  if (blockIdx.z == 0)      { X = q; W = Wq; bias = bq; outp = Qh; }
  else if (blockIdx.z == 1) { X = k; W = Wk; bias = bk; outp = Kh; }
  else                      { X = v; W = Wv; bias = bv; outp = Vh; }

  __shared__ bf16 As[128][40];   // 128 rows x 32 K (pad to 40)
  __shared__ bf16 Bs[128][40];   // 128 N-rows x 32 K (W transposed), pad 40

  const int tid  = threadIdx.x;
  const int m0   = blockIdx.y * 128;
  const int n0   = blockIdx.x * 128;
  const int wave = tid >> 5;
  const int lane = tid & 31;
  const int wm   = wave >> 1;
  const int wn   = wave & 1;
  const int hf   = lane >> 4;
  const int cl   = lane & 15;

  v8f acc[2][4];
#pragma unroll
  for (int i = 0; i < 2; ++i)
#pragma unroll
    for (int j = 0; j < 4; ++j) acc[i][j] = (v8f)(0.0f);

  for (int k0 = 0; k0 < MODEL_DIMS; k0 += 32) {
    __syncthreads();
    {  // stage A: 128x32 fp32 -> bf16, 16 elems/thread
      const int r  = tid >> 1;
      const int cb = (tid & 1) * 16;
      const float* src = X + (size_t)(m0 + r) * MODEL_DIMS + k0 + cb;
      bf16 tmp[16];
#pragma unroll
      for (int i = 0; i < 4; ++i) {
        f32x4 x = *(const f32x4*)(src + i * 4);
#pragma unroll
        for (int e = 0; e < 4; ++e) tmp[i * 4 + e] = (bf16)x[e];
      }
      *(bf16x8*)&As[r][cb]     = *(bf16x8*)&tmp[0];
      *(bf16x8*)&As[r][cb + 8] = *(bf16x8*)&tmp[8];
    }
    {  // stage B transposed: Bs[n][kk] = W[k0+kk][n0+n]
      const int kk = tid >> 3;
      const int nb = (tid & 7) * 16;
      const float* src = W + (size_t)(k0 + kk) * CHANNELS + n0 + nb;
#pragma unroll
      for (int i = 0; i < 4; ++i) {
        f32x4 x = *(const f32x4*)(src + i * 4);
#pragma unroll
        for (int e = 0; e < 4; ++e) Bs[nb + i * 4 + e][kk] = (bf16)x[e];
      }
    }
    __syncthreads();

    v16bf a0 = load_frag(&As[wm * 32][0], 40);
    v16bf a1 = load_frag(&As[wm * 32 + 16][0], 40);
#pragma unroll
    for (int j = 0; j < 4; ++j) {
      v16bf bfr = load_frag(&Bs[wn * 64 + j * 16][0], 40);
      acc[0][j] = wmma_bf16(a0, bfr, acc[0][j]);
      acc[1][j] = wmma_bf16(a1, bfr, acc[1][j]);
    }
  }

#pragma unroll
  for (int i = 0; i < 2; ++i)
#pragma unroll
    for (int j = 0; j < 4; ++j) {
      const int c  = n0 + wn * 64 + j * 16 + cl;
      const float bv_ = bias[c];
      const int h = c >> 7, d = c & 127;
#pragma unroll
      for (int r = 0; r < 8; ++r) {
        const int row  = m0 + wm * 32 + i * 16 + r + 8 * hf;
        const int bidx = row >> 12;
        const int nseq = row & (SEQ - 1);
        const size_t addr =
            (((size_t)bidx * N_HEADS + h) * SEQ + nseq) * HEAD_DIM + d;
        outp[addr] = (bf16)(acc[i][j][r] + bv_);
      }
    }
}

// ---------------------------------------------------------------------------
// Kernel 2: flash attention, double-buffered.  One block per (128 query rows,
// b*h); 8 waves, 16 query rows each.  KV chunk = 64 keys.  K chunks stream
// through the async global->LDS unit (ping-pong buffers, ASYNCcnt waited only
// at the end of each iteration); V chunks are transposed on the way into LDS.
// Dynamic LDS (~88 KB) since CDNA5 allows up to 320 KB per workgroup.
// ---------------------------------------------------------------------------
__global__ __launch_bounds__(256) void flash_attn_kernel(
    const bf16* __restrict__ Qh, const bf16* __restrict__ Kh,
    const bf16* __restrict__ Vh, bf16* __restrict__ Oh) {
  extern __shared__ bf16 smem[];
  bf16* KsB = smem;                       // [2][64][136]
  bf16* VtB = KsB + 2 * 64 * 136;         // [2][128][72]
  bf16* PsB = VtB + 2 * 128 * 72;         // [8][16][72]

  const int tid  = threadIdx.x;
  const int wave = tid >> 5;
  const int lane = tid & 31;
  const int hf   = lane >> 4;
  const int cl   = lane & 15;
  const int bh   = blockIdx.y;                    // 0..63
  const int q0   = blockIdx.x * 128 + wave * 16;  // this wave's query base

  const bf16* Qb = Qh + (size_t)bh * SEQ * HEAD_DIM;
  const bf16* Kb = Kh + (size_t)bh * SEQ * HEAD_DIM;
  const bf16* Vb = Vh + (size_t)bh * SEQ * HEAD_DIM;

  // async K chunk staging: 64x128 bf16, 4 x b128 per thread
  auto stage_k = [&](int kc, int buf) {
    const int r  = tid >> 2;
    const int cb = (tid & 3) * 32;
    const bf16* src = Kb + (size_t)(kc + r) * HEAD_DIM + cb;
#pragma unroll
    for (int i = 0; i < 4; ++i)
      async_load_b128(KsB + ((buf * 64 + r) * 136 + cb + i * 8), src + i * 8);
  };
  // V chunk staged transposed: Vt[d][key]
  auto stage_v = [&](int kc, int buf) {
    const int key = tid >> 2;
    const int db  = (tid & 3) * 32;
    const bf16* src = Vb + (size_t)(kc + key) * HEAD_DIM + db;
#pragma unroll
    for (int i = 0; i < 4; ++i) {
      bf16x8 t = *(const bf16x8*)(src + i * 8);
#pragma unroll
      for (int e = 0; e < 8; ++e)
        VtB[((size_t)buf * 128 + db + i * 8 + e) * 72 + key] = t[e];
    }
  };

  // Q fragments: 16 rows x 128 d -> 4 fragments of 16x32, loaded once.
  v16bf qf[4];
#pragma unroll
  for (int f = 0; f < 4; ++f) {
    const bf16* src = Qb + (size_t)(q0 + cl) * HEAD_DIM + f * 32;
    FragU fr;
    fr.h[0] = *(const bf16x8*)(src + hf * 8);
    fr.h[1] = *(const bf16x8*)(src + 16 + hf * 8);
    qf[f] = fr.v;
  }

  float mrow[8], lrow[8];
  v8f o[8];
#pragma unroll
  for (int j = 0; j < 8; ++j) { mrow[j] = -3.0e38f; lrow[j] = 0.0f; }
#pragma unroll
  for (int t = 0; t < 8; ++t) o[t] = (v8f)(0.0f);

  // prologue: stage chunk 0
  stage_k(0, 0);
  stage_v(0, 0);
  wait_async0();
  __syncthreads();

  const int NCHUNK = SEQ / 64;
  for (int it = 0; it < NCHUNK; ++it) {
    const int kc  = it * 64;
    const int cur = it & 1;
    const int nxt = cur ^ 1;

    // prefetch next K chunk through the async unit while we compute
    if (it + 1 < NCHUNK) stage_k(kc + 64, nxt);

    // S = Q @ K^T : 4 tiles of 16x16 over 64 keys
    v8f st[4];
#pragma unroll
    for (int n = 0; n < 4; ++n) st[n] = (v8f)(0.0f);
#pragma unroll
    for (int n = 0; n < 4; ++n)
#pragma unroll
      for (int dk = 0; dk < 4; ++dk) {
        v16bf bfr = load_frag(KsB + ((cur * 64 + n * 16) * 136 + dk * 32), 136);
        st[n] = wmma_bf16(qf[dk], bfr, st[n]);
      }

    // online softmax (row = vgpr_idx + 8*half; reduce across 16-lane halves)
    float mnew[8];
#pragma unroll
    for (int j = 0; j < 8; ++j) mnew[j] = mrow[j];
#pragma unroll
    for (int n = 0; n < 4; ++n)
#pragma unroll
      for (int j = 0; j < 8; ++j) mnew[j] = fmaxf(mnew[j], st[n][j] * SCALE);
#pragma unroll
    for (int mask = 1; mask <= 8; mask <<= 1)
#pragma unroll
      for (int j = 0; j < 8; ++j)
        mnew[j] = fmaxf(mnew[j], __shfl_xor(mnew[j], mask, 32));

    float alpha[8], rs[8];
#pragma unroll
    for (int j = 0; j < 8; ++j) {
      alpha[j] = __expf(mrow[j] - mnew[j]);
      mrow[j]  = mnew[j];
      rs[j]    = 0.0f;
    }
#pragma unroll
    for (int n = 0; n < 4; ++n)
#pragma unroll
      for (int j = 0; j < 8; ++j) {
        float p = __expf(st[n][j] * SCALE - mnew[j]);
        st[n][j] = p;
        rs[j] += p;
      }
#pragma unroll
    for (int mask = 1; mask <= 8; mask <<= 1)
#pragma unroll
      for (int j = 0; j < 8; ++j) rs[j] += __shfl_xor(rs[j], mask, 32);
#pragma unroll
    for (int j = 0; j < 8; ++j) lrow[j] = lrow[j] * alpha[j] + rs[j];
#pragma unroll
    for (int t = 0; t < 8; ++t)
#pragma unroll
      for (int j = 0; j < 8; ++j) o[t][j] = o[t][j] * alpha[j];

    // prefetch next V chunk (sync transpose) while the async K flies
    if (it + 1 < NCHUNK) stage_v(kc + 64, nxt);

    // P (C layout) -> LDS (row-major) so it re-reads as A fragments
#pragma unroll
    for (int n = 0; n < 4; ++n)
#pragma unroll
      for (int j = 0; j < 8; ++j)
        PsB[((size_t)wave * 16 + j + 8 * hf) * 72 + n * 16 + cl] =
            (bf16)st[n][j];
    __syncthreads();  // uniform; orders Ps write -> read

    // O += P @ V : K-dim = 64 keys (2 A fragments), N = 128 d (8 tiles)
#pragma unroll
    for (int a = 0; a < 2; ++a) {
      v16bf af = load_frag(PsB + ((size_t)wave * 16 * 72 + a * 32), 72);
#pragma unroll
      for (int t = 0; t < 8; ++t) {
        v16bf bfr =
            load_frag(VtB + ((size_t)(cur * 128 + t * 16) * 72 + a * 32), 72);
        o[t] = wmma_bf16(af, bfr, o[t]);
      }
    }

    wait_async0();     // next K chunk resident before anyone consumes it
    __syncthreads();   // buffer swap point
  }

  // epilogue: normalize, store bf16 into [B][SEQ][CHANNELS]
  const int b = bh >> 4, h = bh & 15;
  float rinv[8];
#pragma unroll
  for (int j = 0; j < 8; ++j) rinv[j] = 1.0f / lrow[j];
#pragma unroll
  for (int t = 0; t < 8; ++t)
#pragma unroll
    for (int j = 0; j < 8; ++j) {
      const int n   = q0 + j + 8 * hf;
      const int col = h * HEAD_DIM + t * 16 + cl;
      Oh[((size_t)b * SEQ + n) * CHANNELS + col] = (bf16)(o[t][j] * rinv[j]);
    }
}

// ---------------------------------------------------------------------------
// Kernel 3: output projection.  Oh[16384x2048](bf16) @ Wo[2048x768] + bo,
// fp32 out.  A tiles are pure bf16 copies -> async global->LDS unit.
// ---------------------------------------------------------------------------
__global__ __launch_bounds__(256) void out_proj_kernel(
    const bf16* __restrict__ Oh, const float* __restrict__ Wo,
    const float* __restrict__ bo, float* __restrict__ out) {
  __shared__ bf16 As[128][40];
  __shared__ bf16 Bs[128][40];

  const int tid  = threadIdx.x;
  const int m0   = blockIdx.y * 128;
  const int n0   = blockIdx.x * 128;
  const int wave = tid >> 5;
  const int lane = tid & 31;
  const int wm   = wave >> 1;
  const int wn   = wave & 1;
  const int hf   = lane >> 4;
  const int cl   = lane & 15;

  v8f acc[2][4];
#pragma unroll
  for (int i = 0; i < 2; ++i)
#pragma unroll
    for (int j = 0; j < 4; ++j) acc[i][j] = (v8f)(0.0f);

  for (int k0 = 0; k0 < CHANNELS; k0 += 32) {
    __syncthreads();
    {  // stage A (bf16 straight copy) via async unit
      const int r  = tid >> 1;
      const int cb = (tid & 1) * 16;
      const bf16* src = Oh + (size_t)(m0 + r) * CHANNELS + k0 + cb;
      async_load_b128(&As[r][cb], src);
      async_load_b128(&As[r][cb + 8], src + 8);
    }
    {  // stage B transposed: Bs[n][kk] = Wo[k0+kk][n0+n]
      const int kk = tid >> 3;
      const int nb = (tid & 7) * 16;
      const float* src = Wo + (size_t)(k0 + kk) * MODEL_DIMS + n0 + nb;
#pragma unroll
      for (int i = 0; i < 4; ++i) {
        f32x4 x = *(const f32x4*)(src + i * 4);
#pragma unroll
        for (int e = 0; e < 4; ++e) Bs[nb + i * 4 + e][kk] = (bf16)x[e];
      }
    }
    wait_async0();
    __syncthreads();

    v16bf a0 = load_frag(&As[wm * 32][0], 40);
    v16bf a1 = load_frag(&As[wm * 32 + 16][0], 40);
#pragma unroll
    for (int j = 0; j < 4; ++j) {
      v16bf bfr = load_frag(&Bs[wn * 64 + j * 16][0], 40);
      acc[0][j] = wmma_bf16(a0, bfr, acc[0][j]);
      acc[1][j] = wmma_bf16(a1, bfr, acc[1][j]);
    }
  }

#pragma unroll
  for (int i = 0; i < 2; ++i)
#pragma unroll
    for (int j = 0; j < 4; ++j) {
      const int c = n0 + wn * 64 + j * 16 + cl;
      const float bv_ = bo[c];
#pragma unroll
      for (int r = 0; r < 8; ++r) {
        const int row = m0 + wm * 32 + i * 16 + r + 8 * hf;
        out[(size_t)row * MODEL_DIMS + c] = acc[i][j][r] + bv_;
      }
    }
}

// ---------------------------------------------------------------------------
extern "C" void kernel_launch(void* const* d_in, const int* in_sizes, int n_in,
                              void* d_out, int out_size, void* d_ws,
                              size_t ws_size, hipStream_t stream) {
  const float* q  = (const float*)d_in[0];
  const float* k  = (const float*)d_in[1];
  const float* v  = (const float*)d_in[2];
  const float* Wq = (const float*)d_in[3];
  const float* bq = (const float*)d_in[4];
  const float* Wk = (const float*)d_in[5];
  const float* bk = (const float*)d_in[6];
  const float* Wv = (const float*)d_in[7];
  const float* bv = (const float*)d_in[8];
  const float* Wo = (const float*)d_in[9];
  const float* bo = (const float*)d_in[10];

  const size_t perHeadElems = (size_t)BATCH * N_HEADS * SEQ * HEAD_DIM;
  bf16* Qh = (bf16*)d_ws;
  bf16* Kh = Qh + perHeadElems;
  bf16* Vh = Kh + perHeadElems;
  bf16* Oh = Vh + perHeadElems;  // [B][SEQ][CHANNELS]

  dim3 g1(CHANNELS / 128, (BATCH * SEQ) / 128, 3);  // (16, 128, 3)
  qkv_proj_kernel<<<g1, 256, 0, stream>>>(q, k, v, Wq, bq, Wk, bk, Wv, bv,
                                          Qh, Kh, Vh);

  const size_t flashSmem =
      (size_t)(2 * 64 * 136 + 2 * 128 * 72 + 8 * 16 * 72) * sizeof(bf16);
  dim3 g2(SEQ / 128, BATCH * N_HEADS);              // (32, 64)
  flash_attn_kernel<<<g2, 256, flashSmem, stream>>>(Qh, Kh, Vh, Oh);

  dim3 g3(MODEL_DIMS / 128, (BATCH * SEQ) / 128);   // (6, 128)
  out_proj_kernel<<<g3, 256, 0, stream>>>(Oh, Wo, bo, (float*)d_out);
}
